// TypeAttentionLayer_61641370632922
// MI455X (gfx1250) — compile-verified, matrix operand order checked by reading, not run on hardware
//
#include <hip/hip_runtime.h>
#include <cmath>

#define EPS 1e-8f
#define LOGEPS -18.420680743952367f   // log(1e-8)
#define LDA 36                        // padded half-stride for A tile (18 dwords/row, 8B-aligned rows)

typedef __attribute__((ext_vector_type(16))) _Float16 v16h;
typedef __attribute__((ext_vector_type(8)))  float    v8f;

__global__ void zero_kernel(float* p, int count) {
  int i = blockIdx.x * blockDim.x + threadIdx.x;
  if (i < count) p[i] = 0.f;
}

// Wsum[a, j] = W1[a, j] + W1[a, N + j], stored as f16 (WMMA B operand source)
__global__ void wsum_kernel(const float* __restrict__ W1, _Float16* __restrict__ wsum, int N) {
  int idx = (blockIdx.x * blockDim.x + threadIdx.x) * 4;   // < 128*N
  int a = idx / N, j = idx - a * N;
  const float4 x = *(const float4*)(W1 + (size_t)a * (2 * N) + j);
  const float4 y = *(const float4*)(W1 + (size_t)a * (2 * N) + N + j);
  union { _Float16 h[4]; uint2 u2; } pk;
  pk.h[0] = (_Float16)(x.x + y.x);
  pk.h[1] = (_Float16)(x.y + y.y);
  pk.h[2] = (_Float16)(x.z + y.z);
  pk.h[3] = (_Float16)(x.w + y.w);
  *(uint2*)(wsum + (size_t)a * N + j) = pk.u2;
}

// h = tanh(p_t @ Wsum.T + b1); accumulate column sums of tanh into colsum[t][0..127]
__global__ __launch_bounds__(256) void gemm_scores_kernel(
    const float* __restrict__ p0, const float* __restrict__ p1,
    const float* __restrict__ p2, const float* __restrict__ p3,
    const _Float16* __restrict__ wsum, const float* __restrict__ b1,
    float* __restrict__ colsum, int n, int N, int slabs)
{
  __shared__ _Float16 ldsA[32 * LDA];
  int t    = blockIdx.x / slabs;
  int slab = blockIdx.x - t * slabs;
  const float* p = (t == 0) ? p0 : ((t == 1) ? p1 : ((t == 2) ? p2 : p3));
  int row0 = slab * 32;
  int tid  = threadIdx.x;
  int lane = tid & 31;
  int wave = tid >> 5;
  int lr   = lane & 15;
  int kh2  = (lane >> 4) << 2;          // 0 or 4 (dword units; *2 = half offset 0/8)
  int ccol = wave * 16 + lr;            // output column (A-dim, 0..127)
  const _Float16* brow = wsum + (size_t)ccol * N;
  const unsigned int* ldsU = (const unsigned int*)ldsA;
  // staging coordinates: one float4 per thread per k-step
  int sm = tid >> 3;                    // 0..31 (row of A tile)
  int sk = (tid & 7) << 2;              // 0,4,..,28 (k within tile)
  const float* srow = p + (size_t)(row0 + sm) * N + sk;
  v8f acc0 = {}; v8f acc1 = {};
  for (int k0 = 0; k0 < N; k0 += 32) {
    __syncthreads();
    {
      float4 f = *(const float4*)(srow + k0);
      union { _Float16 h[4]; uint2 u2; } pk;
      pk.h[0] = (_Float16)f.x; pk.h[1] = (_Float16)f.y;
      pk.h[2] = (_Float16)f.z; pk.h[3] = (_Float16)f.w;
      *(uint2*)(ldsA + sm * LDA + sk) = pk.u2;        // ds_store_b64, 8B-aligned
    }
    __syncthreads();
    union { v16h v; unsigned int u[8]; } A0, A1, B;
    #pragma unroll
    for (int i = 0; i < 4; ++i) {
      A0.u[i]     = ldsU[lr * 18 + kh2 + i];          // K = kh + 2i, kh+2i+1
      A0.u[4 + i] = ldsU[lr * 18 + kh2 + 8 + i];      // K = kh + 16 + 2i
      A1.u[i]     = ldsU[(16 + lr) * 18 + kh2 + i];
      A1.u[4 + i] = ldsU[(16 + lr) * 18 + kh2 + 8 + i];
      B.u[i]      = *(const unsigned int*)(brow + k0 + (kh2 << 1) + 2 * i);
      B.u[4 + i]  = *(const unsigned int*)(brow + k0 + (kh2 << 1) + 16 + 2 * i);
    }
    acc0 = __builtin_amdgcn_wmma_f32_16x16x32_f16(false, A0.v, false, B.v,
                                                  (short)0, acc0, false, false);
    acc1 = __builtin_amdgcn_wmma_f32_16x16x32_f16(false, A1.v, false, B.v,
                                                  (short)0, acc1, false, false);
  }
  float bv = b1[ccol];
  float partial = 0.f;
  #pragma unroll
  for (int r = 0; r < 8; ++r)
    partial += tanhf(acc0[r] + bv) + tanhf(acc1[r] + bv);
  atomicAdd(&colsum[t * 128 + ccol], partial);
}

// scores -> softmax -> mixing scalars (bu, ub, bs, sb)
__global__ void score_softmax_kernel(const float* __restrict__ colsum,
                                     const float* __restrict__ W2,
                                     float* __restrict__ scal, int n) {
  __shared__ float red[4][128];
  int c = threadIdx.x;                  // 128 threads
  for (int t = 0; t < 4; ++t)
    red[t][c] = (colsum[t * 128 + c] / (float)n) * W2[c];
  __syncthreads();
  if (c == 0) {
    float s[4];
    for (int t = 0; t < 4; ++t) { float a = 0.f; for (int i = 0; i < 128; ++i) a += red[t][i]; s[t] = a; }
    float m = fmaxf(fmaxf(s[0], s[1]), fmaxf(s[2], s[3]));
    float e[4], tot = 0.f;
    for (int t = 0; t < 4; ++t) { e[t] = expf(s[t] - m); tot += e[t]; }
    float a0 = e[0] / tot, a1 = e[1] / tot, a2 = e[2] / tot;
    scal[0] = a0 / (a0 + a1);  // bu
    scal[1] = a1 / (a0 + a1);  // ub
    scal[2] = a0 / (a0 + a2);  // bs
    scal[3] = a2 / (a0 + a2);  // sb
  }
}

// out[j..j+3] = sum_i p[i, colOff + j..j+3], 4 columns per thread
__global__ void csum_kernel(const float* __restrict__ p, int colOff, int n, int N,
                            float* __restrict__ out) {
  int j4 = (blockIdx.x * blockDim.x + threadIdx.x) * 4;
  if (j4 >= n) return;
  float4 s = make_float4(0.f, 0.f, 0.f, 0.f);
  for (int i = 0; i < n; ++i) {
    float4 v = *(const float4*)(p + (size_t)i * N + colOff + j4);
    s.x += v.x; s.y += v.y; s.z += v.z; s.w += v.w;
  }
  *(float4*)(out + j4) = s;
}

// out[row] = sum_k p[row, colOff + k], k < n   (one wave per row, float4 per lane)
__global__ void rsum_kernel(const float* __restrict__ p, int colOff, int n, int N,
                            float* __restrict__ out) {
  int wave = threadIdx.x >> 5, lane = threadIdx.x & 31;
  int row = blockIdx.x * 8 + wave;
  if (row >= n) return;
  const float* pr = p + (size_t)row * N + colOff;
  float4 a = make_float4(0.f, 0.f, 0.f, 0.f);
  for (int k = lane * 4; k < n; k += 128) {
    float4 v = *(const float4*)(pr + k);
    a.x += v.x; a.y += v.y; a.z += v.z; a.w += v.w;
  }
  float s = (a.x + a.y) + (a.z + a.w);
  for (int off = 16; off > 0; off >>= 1) s += __shfl_xor(s, off, 32);
  if (lane == 0) out[row] = s;
}

// rowsum[r] of the symmetrized M matrix, from the 12 partial-sum vectors
__global__ void rowsum_combine_kernel(const float* __restrict__ vec,
                                      const float* __restrict__ scal,
                                      float* __restrict__ rowsum, int n) {
  int r = blockIdx.x * blockDim.x + threadIdx.x;
  if (r >= 4 * n) return;
  const float *c0a = vec,         *c0b = vec + n,     *c0c = vec + 2 * n;
  const float *c1  = vec + 3 * n, *c2  = vec + 4 * n, *c3  = vec + 5 * n;
  const float *r0a = vec + 6 * n, *r0b = vec + 7 * n, *r0c = vec + 8 * n;
  const float *r1  = vec + 9 * n, *r2  = vec + 10 * n,*r3  = vec + 11 * n;
  float bu = scal[0], ub = scal[1], bs = scal[2], sb = scal[3];
  float rs;
  if (r < n)            rs = bu * r0a[r] + ub * c1[r] + bs * (r0b[r] + r0c[r]) + sb * (c2[r] + c3[r]);
  else if (r < 2 * n) { int j = r - n;     rs = bu * c0a[j] + ub * r1[j]; }
  else if (r < 3 * n) { int j = r - 2 * n; rs = bs * c0b[j] + sb * r2[j]; }
  else                { int j = r - 3 * n; rs = bs * c0c[j] + sb * r3[j]; }
  rowsum[r] = rs;
}

// final [N x N] output: 64x64 tiles; compute tiles also write their mirror tile
__global__ __launch_bounds__(256) void output_kernel(
    const float* __restrict__ p0, const float* __restrict__ p1,
    const float* __restrict__ p2, const float* __restrict__ p3,
    const float* __restrict__ scal, const float* __restrict__ rowsum,
    float* __restrict__ out, int n)
{
  const int N = 4 * n;
  __shared__ float lds0[64 * 65];
  __shared__ float ldsT[64 * 65];
  int i0 = blockIdx.y * 64;
  int j0 = blockIdx.x * 64;
  int tid = threadIdx.x;
  int tr4 = tid >> 4;           // 0..15 (row group)
  int tc4 = (tid & 15) * 4;     // 0,4,..,60 (4-wide column chunk)
  bool topRows = (i0 < n), leftCols = (j0 < n);
  if (!topRows && leftCols) return;            // handled by mirror of a compute tile
  if (topRows == leftCols) {                   // structurally-zero regions
    const float4 o = make_float4(LOGEPS, LOGEPS, LOGEPS, LOGEPS);
    #pragma unroll
    for (int q = 0; q < 4; ++q) {
      int rr = tr4 + q * 16;
      *(float4*)(out + (size_t)(i0 + rr) * N + j0 + tc4) = o;
    }
    return;
  }
  int blk = j0 / n;                            // 1, 2 or 3
  const float* pt = (blk == 1) ? p1 : ((blk == 2) ? p2 : p3);
  float s0 = (blk == 1) ? scal[0] : scal[2];   // bu or bs
  float s1 = (blk == 1) ? scal[1] : scal[3];   // ub or sb
  #pragma unroll
  for (int q = 0; q < 4; ++q) {                // stage both 64x64 tiles (float4 loads)
    int jj = tr4 + q * 16;
    float4 a = *(const float4*)(p0 + (size_t)(i0 + jj) * N + j0 + tc4);
    lds0[jj * 65 + tc4 + 0] = a.x; lds0[jj * 65 + tc4 + 1] = a.y;
    lds0[jj * 65 + tc4 + 2] = a.z; lds0[jj * 65 + tc4 + 3] = a.w;
    float4 b = *(const float4*)(pt + (size_t)(j0 - blk * n + jj) * N + i0 + tc4);
    ldsT[jj * 65 + tc4 + 0] = b.x; ldsT[jj * 65 + tc4 + 1] = b.y;
    ldsT[jj * 65 + tc4 + 2] = b.z; ldsT[jj * 65 + tc4 + 3] = b.w;
  }
  __syncthreads();
  #pragma unroll
  for (int q = 0; q < 4; ++q) {                // top tile (rows i0.., cols j0..)
    int rr = tr4 + q * 16;
    float inv = 1.f / (rowsum[i0 + rr] + EPS);
    float4 o;
    o.x = logf((s0 * lds0[rr * 65 + tc4 + 0] + s1 * ldsT[(tc4 + 0) * 65 + rr]) * inv + EPS);
    o.y = logf((s0 * lds0[rr * 65 + tc4 + 1] + s1 * ldsT[(tc4 + 1) * 65 + rr]) * inv + EPS);
    o.z = logf((s0 * lds0[rr * 65 + tc4 + 2] + s1 * ldsT[(tc4 + 2) * 65 + rr]) * inv + EPS);
    o.w = logf((s0 * lds0[rr * 65 + tc4 + 3] + s1 * ldsT[(tc4 + 3) * 65 + rr]) * inv + EPS);
    *(float4*)(out + (size_t)(i0 + rr) * N + j0 + tc4) = o;
  }
  #pragma unroll
  for (int q = 0; q < 4; ++q) {                // mirror tile (rows j0.., cols i0..)
    int jj = tr4 + q * 16;
    float inv = 1.f / (rowsum[j0 + jj] + EPS);
    float4 o;
    o.x = logf((s0 * lds0[(tc4 + 0) * 65 + jj] + s1 * ldsT[jj * 65 + tc4 + 0]) * inv + EPS);
    o.y = logf((s0 * lds0[(tc4 + 1) * 65 + jj] + s1 * ldsT[jj * 65 + tc4 + 1]) * inv + EPS);
    o.z = logf((s0 * lds0[(tc4 + 2) * 65 + jj] + s1 * ldsT[jj * 65 + tc4 + 2]) * inv + EPS);
    o.w = logf((s0 * lds0[(tc4 + 3) * 65 + jj] + s1 * ldsT[jj * 65 + tc4 + 3]) * inv + EPS);
    *(float4*)(out + (size_t)(j0 + jj) * N + i0 + tc4) = o;
  }
}

extern "C" void kernel_launch(void* const* d_in, const int* in_sizes, int n_in,
                              void* d_out, int out_size, void* d_ws, size_t ws_size,
                              hipStream_t stream) {
  (void)n_in; (void)out_size; (void)ws_size;
  const float* p0 = (const float*)d_in[0];
  const float* p1 = (const float*)d_in[1];
  const float* p2 = (const float*)d_in[2];
  const float* p3 = (const float*)d_in[3];
  const float* W1 = (const float*)d_in[4];
  const float* b1 = (const float*)d_in[5];
  const float* W2 = (const float*)d_in[6];
  int n = (int)(sqrt((double)in_sizes[0] * 0.25) + 0.5);   // prob0 is [n, 4n]
  int N = 4 * n;

  // workspace layout (bytes)
  char* ws = (char*)d_ws;
  float*    colsum = (float*)ws;                                   // 512 f32
  float*    scal   = (float*)(ws + 2048);                          // 4 f32
  _Float16* wsum   = (_Float16*)(ws + 4096);                       // 128*N f16
  float*    vec    = (float*)(ws + 4096 + (size_t)128 * N * 2);    // 12*n f32
  float*    rowsum = vec + 12 * n;                                 // 4*n f32

  zero_kernel<<<2, 256, 0, stream>>>(colsum, 512);
  wsum_kernel<<<(128 * N) / 1024, 256, 0, stream>>>(W1, wsum, N);
  int slabs = n / 32;
  gemm_scores_kernel<<<4 * slabs, 256, 0, stream>>>(p0, p1, p2, p3, wsum, b1,
                                                    colsum, n, N, slabs);
  score_softmax_kernel<<<1, 128, 0, stream>>>(colsum, W2, scal, n);

  int cb = (n / 4 + 255) / 256, rb = (n + 7) / 8;
  csum_kernel<<<cb, 256, 0, stream>>>(p0, n,     n, N, vec + 0 * n);
  csum_kernel<<<cb, 256, 0, stream>>>(p0, 2 * n, n, N, vec + 1 * n);
  csum_kernel<<<cb, 256, 0, stream>>>(p0, 3 * n, n, N, vec + 2 * n);
  csum_kernel<<<cb, 256, 0, stream>>>(p1, 0,     n, N, vec + 3 * n);
  csum_kernel<<<cb, 256, 0, stream>>>(p2, 0,     n, N, vec + 4 * n);
  csum_kernel<<<cb, 256, 0, stream>>>(p3, 0,     n, N, vec + 5 * n);
  rsum_kernel<<<rb, 256, 0, stream>>>(p0, n,     n, N, vec + 6 * n);
  rsum_kernel<<<rb, 256, 0, stream>>>(p0, 2 * n, n, N, vec + 7 * n);
  rsum_kernel<<<rb, 256, 0, stream>>>(p0, 3 * n, n, N, vec + 8 * n);
  rsum_kernel<<<rb, 256, 0, stream>>>(p1, 0,     n, N, vec + 9 * n);
  rsum_kernel<<<rb, 256, 0, stream>>>(p2, 0,     n, N, vec + 10 * n);
  rsum_kernel<<<rb, 256, 0, stream>>>(p3, 0,     n, N, vec + 11 * n);
  rowsum_combine_kernel<<<(4 * n + 255) / 256, 256, 0, stream>>>(vec, scal, rowsum, n);

  dim3 g2(N / 64, N / 64);
  output_kernel<<<g2, 256, 0, stream>>>(p0, p1, p2, p3, scal, rowsum,
                                        (float*)d_out, n);
}
